// InferenceCLT_87239375716481
// MI455X (gfx1250) — matrix operand order repaired and checked by reading, not compiled, not run on hardware
//
#include <hip/hip_runtime.h>

#define T_TOK 2048
#define NL 12
#define DM 768
#define FL 2048

typedef __bf16 bf16;
typedef bf16  v2bf  __attribute__((ext_vector_type(2)));
typedef bf16  v16bf __attribute__((ext_vector_type(16)));
typedef float v8f   __attribute__((ext_vector_type(8)));

union Frag {
    v16bf v;
    unsigned int u[8];
    uint4 q[2];
};

// Hardware RNE converts (v_cvt_*bf16_f32) instead of bit-twiddling.
__device__ __forceinline__ unsigned short f2bf(float x) {
    return __builtin_bit_cast(unsigned short, (bf16)x);
}
__device__ __forceinline__ unsigned int pack2(float a, float b) {
    v2bf p;
    p.x = (bf16)a;
    p.y = (bf16)b;
    return __builtin_bit_cast(unsigned int, p);
}

// LDS tiles: [rows][32] bf16, row stride 40 ushorts (80 B). 80 % 16 == 0 so
// every fragment read is a 16B-aligned ds_load_b128; +8 pad staggers banks.
#define LDS_STRIDE 40

// ---------------------------------------------------------------------------
// Encode + JumpReLU:  h[t,l,f] = sum_d x[t,l,d] * w_e[l,d,f];  acts = h*(h>e^t)
// grid = (T/128, F/128, L), block = 256 (8 waves), wave w owns rows w*16..+16
// ---------------------------------------------------------------------------
__global__ __launch_bounds__(256) void clt_encode(
    const float* __restrict__ x, const float* __restrict__ w_e,
    const float* __restrict__ tlog, float* __restrict__ h_out,
    float* __restrict__ a_out, unsigned short* __restrict__ a_bf) {
    __shared__ __align__(16) unsigned short As[128 * LDS_STRIDE];
    __shared__ __align__(16) unsigned short Bs[128 * LDS_STRIDE];
    const int tid = threadIdx.x;
    const int t0 = blockIdx.x * 128;
    const int f0 = blockIdx.y * 128;
    const int l  = blockIdx.z;
    const int wave = tid >> 5, lane = tid & 31, lh = lane >> 4, ln = lane & 15;

    v8f acc[8];
#pragma unroll
    for (int n = 0; n < 8; ++n)
#pragma unroll
        for (int r = 0; r < 8; ++r) acc[n][r] = 0.f;

    // Register prefetch buffers (software pipeline over k).
    float4 pa[4];          // A: 128 rows x 32 cols (8 float4 per row)
    float4 pbA[2], pbB[2]; // B: 32 d-rows x 128 f-cols, as 16 row-PAIRS x 32 f4

    auto loadA = [&](int k0) {
#pragma unroll
        for (int it = 0; it < 4; ++it) {
            int idx = it * 256 + tid;
            int r = idx >> 3, cv = idx & 7;
            pa[it] = *(const float4*)(x + (((size_t)(t0 + r) * NL + l) * DM + k0 + cv * 4));
        }
    };
    auto loadB = [&](int k0) {
#pragma unroll
        for (int it = 0; it < 2; ++it) {
            int idx = it * 256 + tid;
            int rp = idx >> 5, fv = idx & 31;   // rows d = k0+2rp, k0+2rp+1
            const float* base = w_e + (((size_t)l * DM + k0 + 2 * rp) * FL + f0 + fv * 4);
            pbA[it] = *(const float4*)base;
            pbB[it] = *(const float4*)(base + FL);
        }
    };
    auto storeTiles = [&]() {
#pragma unroll
        for (int it = 0; it < 4; ++it) {
            int idx = it * 256 + tid;
            int r = idx >> 3, cv = idx & 7;
            unsigned int* dst = (unsigned int*)&As[r * LDS_STRIDE + cv * 4];
            dst[0] = pack2(pa[it].x, pa[it].y);
            dst[1] = pack2(pa[it].z, pa[it].w);
        }
#pragma unroll
        for (int it = 0; it < 2; ++it) {
            int idx = it * 256 + tid;
            int rp = idx >> 5, fv = idx & 31;
            const float* fa = (const float*)&pbA[it];
            const float* fb = (const float*)&pbB[it];
#pragma unroll
            for (int q = 0; q < 4; ++q) {
                // Bs[f][d]: packed (d=2rp lo, d=2rp+1 hi) -> one b32 store.
                *(unsigned int*)&Bs[(fv * 4 + q) * LDS_STRIDE + 2 * rp] =
                    pack2(fa[q], fb[q]);
            }
        }
    };

    loadA(0);
    loadB(0);
    const int nsteps = DM / 32;  // 24
    for (int s = 0; s < nsteps; ++s) {
        __syncthreads();
        storeTiles();
        if (s + 1 < nsteps) { loadA((s + 1) * 32); loadB((s + 1) * 32); }
        __syncthreads();

        Frag a;
        const unsigned short* Arow = &As[(wave * 16 + ln) * LDS_STRIDE];
        a.q[0] = *(const uint4*)&Arow[lh * 8];          // K = 8h .. +7
        a.q[1] = *(const uint4*)&Arow[16 + lh * 8];     // K = 16+8h .. +7
#pragma unroll
        for (int nt = 0; nt < 8; ++nt) {
            Frag b;
            const unsigned short* Brow = &Bs[(nt * 16 + ln) * LDS_STRIDE];
            b.q[0] = *(const uint4*)&Brow[lh * 8];
            b.q[1] = *(const uint4*)&Brow[16 + lh * 8];
            acc[nt] = __builtin_amdgcn_wmma_f32_16x16x32_bf16(
                false, a.v, false, b.v, (short)0, acc[nt], false, false);
        }
    }

    // Epilogue: JumpReLU; branch on a_bf hoisted out of the store loops.
    if (a_bf) {
#pragma unroll
        for (int nt = 0; nt < 8; ++nt) {
            int f = f0 + nt * 16 + ln;
            float thr = __expf(tlog[l * FL + f]);
#pragma unroll
            for (int r = 0; r < 8; ++r) {
                int row = t0 + wave * 16 + lh * 8 + r;
                size_t oi = ((size_t)row * NL + l) * FL + f;
                float hv = acc[nt][r];
                float av = hv > thr ? hv : 0.f;
                h_out[oi] = hv;
                a_out[oi] = av;
                a_bf[oi] = f2bf(av);
            }
        }
    } else {
#pragma unroll
        for (int nt = 0; nt < 8; ++nt) {
            int f = f0 + nt * 16 + ln;
            float thr = __expf(tlog[l * FL + f]);
#pragma unroll
            for (int r = 0; r < 8; ++r) {
                int row = t0 + wave * 16 + lh * 8 + r;
                size_t oi = ((size_t)row * NL + l) * FL + f;
                float hv = acc[nt][r];
                h_out[oi] = hv;
                a_out[oi] = hv > thr ? hv : 0.f;
            }
        }
    }
}

// ---------------------------------------------------------------------------
// Triangular decode: x_hat[t,j,d] = sum_{i<=j} sum_f acts[t,i,f]*w_d[i,j,f,d]
// grid = (T/256, D/128, 12), j = 11 - blockIdx.z (largest K first).
// block = 256 (8 waves); wave owns a 32x128 strip (2 A frags x 8 B frags
// -> 16 WMMAs per 20 ds_load_b128).
// ---------------------------------------------------------------------------
template<bool USE_BF>
__global__ __launch_bounds__(256) void clt_decode(
    const float* __restrict__ acts_f, const unsigned short* __restrict__ acts_bf,
    const float* __restrict__ w_d, float* __restrict__ xhat) {
    __shared__ __align__(16) unsigned short As[256 * LDS_STRIDE];
    __shared__ __align__(16) unsigned short Bs[128 * LDS_STRIDE];
    const int tid = threadIdx.x;
    const int t0 = blockIdx.x * 256;
    const int d0 = blockIdx.y * 128;
    const int j  = (NL - 1) - (int)blockIdx.z;
    const int wave = tid >> 5, lane = tid & 31, lh = lane >> 4, ln = lane & 15;

    v8f acc0[8], acc1[8];
#pragma unroll
    for (int n = 0; n < 8; ++n)
#pragma unroll
        for (int r = 0; r < 8; ++r) { acc0[n][r] = 0.f; acc1[n][r] = 0.f; }

    const int KSTEPS = FL / 32;            // 64 k-steps per source layer i
    const int nsteps = (j + 1) * KSTEPS;

    // Register prefetch buffers.
    uint4  pa_bf[4];       // A tile (bf16 ws path): 256 rows x 32 cols
    float4 pa_f[8];        // A tile (fp32 fallback)
    float4 pbA[2], pbB[2]; // B: 32 f-rows x 128 d-cols as 16 row-pairs x 32 d4

    auto loadA = [&](int s) {
        int i = s >> 6, k0 = (s & (KSTEPS - 1)) << 5;
        if (USE_BF) {
#pragma unroll
            for (int it = 0; it < 4; ++it) {
                int idx = it * 256 + tid;
                int r = idx >> 2, cv = idx & 3;
                pa_bf[it] = *(const uint4*)(acts_bf + (((size_t)(t0 + r) * NL + i) * FL + k0 + cv * 8));
            }
        } else {
#pragma unroll
            for (int it = 0; it < 8; ++it) {
                int idx = it * 256 + tid;
                int r = idx >> 3, cv = idx & 7;
                pa_f[it] = *(const float4*)(acts_f + (((size_t)(t0 + r) * NL + i) * FL + k0 + cv * 4));
            }
        }
    };
    auto loadB = [&](int s) {
        int i = s >> 6, k0 = (s & (KSTEPS - 1)) << 5;
#pragma unroll
        for (int it = 0; it < 2; ++it) {
            int idx = it * 256 + tid;
            int rp = idx >> 5, dv = idx & 31;   // rows f = k0+2rp, k0+2rp+1
            const float* base = w_d + ((((size_t)i * NL + j) * FL + k0 + 2 * rp) * DM + d0 + dv * 4);
            pbA[it] = *(const float4*)base;
            pbB[it] = *(const float4*)(base + DM);
        }
    };
    auto storeTiles = [&]() {
        if (USE_BF) {
#pragma unroll
            for (int it = 0; it < 4; ++it) {
                int idx = it * 256 + tid;
                int r = idx >> 2, cv = idx & 3;
                *(uint4*)&As[r * LDS_STRIDE + cv * 8] = pa_bf[it];
            }
        } else {
#pragma unroll
            for (int it = 0; it < 8; ++it) {
                int idx = it * 256 + tid;
                int r = idx >> 3, cv = idx & 7;
                unsigned int* dst = (unsigned int*)&As[r * LDS_STRIDE + cv * 4];
                dst[0] = pack2(pa_f[it].x, pa_f[it].y);
                dst[1] = pack2(pa_f[it].z, pa_f[it].w);
            }
        }
#pragma unroll
        for (int it = 0; it < 2; ++it) {
            int idx = it * 256 + tid;
            int rp = idx >> 5, dv = idx & 31;
            const float* fa = (const float*)&pbA[it];
            const float* fb = (const float*)&pbB[it];
#pragma unroll
            for (int q = 0; q < 4; ++q) {
                // Bs[d][f]: packed (f=2rp lo, f=2rp+1 hi) -> one b32 store.
                *(unsigned int*)&Bs[(dv * 4 + q) * LDS_STRIDE + 2 * rp] =
                    pack2(fa[q], fb[q]);
            }
        }
    };

    loadA(0);
    loadB(0);
    for (int s = 0; s < nsteps; ++s) {
        __syncthreads();
        storeTiles();
        if (s + 1 < nsteps) { loadA(s + 1); loadB(s + 1); }  // overlap HBM w/ WMMA
        __syncthreads();

        Frag a0, a1;
        const unsigned short* Ar0 = &As[(wave * 32 + ln) * LDS_STRIDE];
        const unsigned short* Ar1 = &As[(wave * 32 + 16 + ln) * LDS_STRIDE];
        a0.q[0] = *(const uint4*)&Ar0[lh * 8];
        a0.q[1] = *(const uint4*)&Ar0[16 + lh * 8];
        a1.q[0] = *(const uint4*)&Ar1[lh * 8];
        a1.q[1] = *(const uint4*)&Ar1[16 + lh * 8];
#pragma unroll
        for (int nt = 0; nt < 8; ++nt) {
            Frag b;
            const unsigned short* Brow = &Bs[(nt * 16 + ln) * LDS_STRIDE];
            b.q[0] = *(const uint4*)&Brow[lh * 8];
            b.q[1] = *(const uint4*)&Brow[16 + lh * 8];
            acc0[nt] = __builtin_amdgcn_wmma_f32_16x16x32_bf16(
                false, a0.v, false, b.v, (short)0, acc0[nt], false, false);
            acc1[nt] = __builtin_amdgcn_wmma_f32_16x16x32_bf16(
                false, a1.v, false, b.v, (short)0, acc1[nt], false, false);
        }
    }

#pragma unroll
    for (int nt = 0; nt < 8; ++nt) {
        int d = d0 + nt * 16 + ln;
#pragma unroll
        for (int r = 0; r < 8; ++r) {
            int row0 = t0 + wave * 32 + lh * 8 + r;
            xhat[((size_t)row0 * NL + j) * DM + d] = acc0[nt][r];
            xhat[((size_t)(row0 + 16) * NL + j) * DM + d] = acc1[nt][r];
        }
    }
}

extern "C" void kernel_launch(void* const* d_in, const int* in_sizes, int n_in,
                              void* d_out, int out_size, void* d_ws, size_t ws_size,
                              hipStream_t stream) {
    const float* x   = (const float*)d_in[0];   // [T, L, D]
    const float* w_e = (const float*)d_in[1];   // [L, D, F]
    const float* w_d = (const float*)d_in[2];   // [L, L, F, D]
    const float* t   = (const float*)d_in[3];   // [L, F]

    float* out = (float*)d_out;
    const size_t hf_elems = (size_t)T_TOK * NL * FL;   // h and acts each
    float* h_out    = out;
    float* acts_out = out + hf_elems;
    float* xhat_out = out + 2 * hf_elems;

    unsigned short* acts_bf = nullptr;
    if (ws_size >= hf_elems * sizeof(unsigned short))
        acts_bf = (unsigned short*)d_ws;

    dim3 blk(256, 1, 1);
    dim3 ge(T_TOK / 128, FL / 128, NL);   // (16, 16, 12)
    clt_encode<<<ge, blk, 0, stream>>>(x, w_e, t, h_out, acts_out, acts_bf);

    dim3 gd(T_TOK / 256, DM / 128, NL);   // (8, 6, 12)
    if (acts_bf)
        clt_decode<true><<<gd, blk, 0, stream>>>(acts_out, acts_bf, w_d, xhat_out);
    else
        clt_decode<false><<<gd, blk, 0, stream>>>(acts_out, acts_bf, w_d, xhat_out);
}
